// HexInpaintingHead_17394617548922
// MI455X (gfx1250) — matrix operand order repaired and checked by reading, not compiled, not run on hardware
//
#include <hip/hip_runtime.h>
#include <hip/hip_bf16.h>

// ---------------------------------------------------------------------------
// HexInpaintingHead on MI455X (gfx1250): bf16 WMMA token MLPs with weights
// pre-packed into B-fragment order, async global->LDS staging (ASYNCcnt),
// f32 atomics for edge scatter, per-batch scan for masked compaction.
// ---------------------------------------------------------------------------

#define B_SZ   128
#define N_SZ   4096
#define H_SZ   96
#define H4_SZ  384
#define E_SZ   24576
#define LAT_SZ 1024
#define OUT_CH 2
#define LN_EPS 1e-5f

typedef __attribute__((ext_vector_type(16))) __bf16          v16bf;
typedef __attribute__((ext_vector_type(16))) unsigned short  v16u;
typedef __attribute__((ext_vector_type(8)))  unsigned short  v8u;
typedef __attribute__((ext_vector_type(8)))  float           v8f;
typedef __attribute__((ext_vector_type(4)))  int             v4i;

#if defined(__has_builtin)
#if __has_builtin(__builtin_amdgcn_global_load_async_to_lds_b128) && \
    __has_builtin(__builtin_amdgcn_s_wait_asynccnt)
#define HAVE_ASYNC_LDS 1
#endif
#endif

#define GLOBAL_AS __attribute__((address_space(1)))
#define LDS_AS    __attribute__((address_space(3)))

// float -> bf16 bits, round-to-nearest-even (no scalar __bf16 arithmetic)
static __device__ __forceinline__ unsigned short f2bf(float f) {
    unsigned int u = __float_as_uint(f);
    unsigned int r = (u + 0x7FFFu + ((u >> 16) & 1u)) >> 16;
    return (unsigned short)r;
}

static __device__ __forceinline__ float gelu_exact(float v) {
    return 0.5f * v * (1.0f + erff(v * 0.70710678118654752f));
}

// A-fragment from an LDS row (16-bit 16x32 layout): lane<16 -> K{k0..k0+7,
// k0+16..k0+23}; lane>=16 -> K{k0+8..15, k0+24..31}. Row must be 16B aligned.
static __device__ __forceinline__ v16bf lds_frag_a(const unsigned short* row, int k0, int hi) {
    const v8u* rp = (const v8u*)row;
    v8u lo = rp[(k0 >> 3) + hi];            // k0 + hi*8 .. +7        (ds_load_b128)
    v8u hh = rp[(k0 >> 3) + 2 + hi];        // k0 + 16 + hi*8 .. +7   (ds_load_b128)
    v16u full = __builtin_shufflevector(lo, hh, 0, 1, 2, 3, 4, 5, 6, 7,
                                                8, 9, 10, 11, 12, 13, 14, 15);
    return __builtin_bit_cast(v16bf, full);
}

// ---------------------------------------------------------------------------
// utility kernels
// ---------------------------------------------------------------------------
__global__ void zero_f32(float* p, long n) {
    long i = (long)blockIdx.x * blockDim.x + threadIdx.x;
    long stride = (long)gridDim.x * blockDim.x;
    for (; i < n; i += stride) p[i] = 0.0f;
}

__global__ void zero_bytes(unsigned char* p, long n) {
    long i = (long)blockIdx.x * blockDim.x + threadIdx.x;
    long stride = (long)gridDim.x * blockDim.x;
    for (; i < n; i += stride) p[i] = 0;
}

// ---------------------------------------------------------------------------
// Pack f32 weights (K x N, leading dim ldn) into bf16 WMMA B-fragment order:
// dst[((nt*kchunks + kc)*32 + lane)*16 + i] =
//     bf16( w[(kc*32 + (lane>>4)*8 + (i&7) + (i>>3)*16)*ldn + nt*16 + (lane&15)] )
// so each lane's 16-half fragment is one contiguous 32-byte load.
// ---------------------------------------------------------------------------
__global__ void pack_frag_kernel(const float* __restrict__ w, unsigned short* __restrict__ dst,
                                 int ldn, int ntiles, int kchunks) {
    int idx = blockIdx.x * blockDim.x + threadIdx.x;    // (nt*kchunks + kc)*32 + lane
    if (idx >= ntiles * kchunks * 32) return;
    int lane = idx & 31;
    int fc   = idx >> 5;
    int kc   = fc % kchunks;
    int nt   = fc / kchunks;
    int kb   = kc * 32 + (lane >> 4) * 8;
    int n    = nt * 16 + (lane & 15);
    v16u v;
    #pragma unroll
    for (int i = 0; i < 16; ++i) v[i] = f2bf(w[(kb + (i & 7) + ((i >> 3) << 4)) * ldn + n]);
    *(v16u*)(dst + (long)idx * 16) = v;
}

// ---------------------------------------------------------------------------
// lat = gelu(latent_token @ lat_w + lat_b)       one thread per (b,h)
// ---------------------------------------------------------------------------
__global__ void latent_kernel(const float* __restrict__ tok, const float* __restrict__ w,
                              const float* __restrict__ b, float* __restrict__ lat) {
    int i = blockIdx.x * blockDim.x + threadIdx.x;      // b*H + h
    if (i >= B_SZ * H_SZ) return;
    int bb = i / H_SZ, h = i % H_SZ;
    const float* t = tok + (long)bb * LAT_SZ;
    float acc = b[h];
    for (int k = 0; k < LAT_SZ; ++k) acc += t[k] * w[k * H_SZ + h];
    lat[i] = gelu_exact(acc);
}

// ---------------------------------------------------------------------------
// x = concat(nf, lat) @ in_w + in_b              one thread per (b,n,h)
// ---------------------------------------------------------------------------
__global__ void input_proj_kernel(const float* __restrict__ nf, const float* __restrict__ lat,
                                  const float* __restrict__ w, const float* __restrict__ b,
                                  float* __restrict__ x) {
    long i = (long)blockIdx.x * blockDim.x + threadIdx.x;
    long total = (long)B_SZ * N_SZ * H_SZ;
    if (i >= total) return;
    int h = (int)(i % H_SZ);
    long tokid = i / H_SZ;                  // b*N + n
    int bb = (int)(tokid / N_SZ);
    float acc = b[h];
    acc += nf[tokid * 2 + 0] * w[0 * H_SZ + h];
    acc += nf[tokid * 2 + 1] * w[1 * H_SZ + h];
    const float* lr = lat + (long)bb * H_SZ;
    for (int k = 0; k < H_SZ; ++k) acc += lr[k] * w[(2 + k) * H_SZ + h];
    x[i] = acc;
}

// ---------------------------------------------------------------------------
// degree + inverse
// ---------------------------------------------------------------------------
__global__ void deg_kernel(const long long* __restrict__ ei, float* __restrict__ deg) {
    int e = blockIdx.x * blockDim.x + threadIdx.x;
    if (e < E_SZ) atomicAdd(&deg[(int)ei[E_SZ + e]], 1.0f);     // dst row
}

__global__ void inv_kernel(const float* __restrict__ deg, float* __restrict__ inv) {
    int n = blockIdx.x * blockDim.x + threadIdx.x;
    if (n < N_SZ) inv[n] = 1.0f / (deg[n] + 1.0f);
}

// ---------------------------------------------------------------------------
// agg[:, dst, :] += x[:, src, :]                 one thread per (b,e)
// ---------------------------------------------------------------------------
__global__ void scatter_kernel(const float* __restrict__ x, const long long* __restrict__ ei,
                               float* __restrict__ agg) {
    long i = (long)blockIdx.x * blockDim.x + threadIdx.x;
    if (i >= (long)B_SZ * E_SZ) return;
    int bb = (int)(i / E_SZ), e = (int)(i % E_SZ);
    int s = (int)ei[e];
    int d = (int)ei[E_SZ + e];
    const float* xr = x + ((long)bb * N_SZ + s) * H_SZ;
    float* ar = agg + ((long)bb * N_SZ + d) * H_SZ;
    #pragma unroll
    for (int c = 0; c < H_SZ; ++c) atomicAdd(&ar[c], xr[c]);
}

// ---------------------------------------------------------------------------
// Block body: h=(agg+x)*inv; LN; gelu(h@w1+b1)@w2+b2; x += gamma*h
// 16 tokens per workgroup, 8 waves, bf16 WMMA 16x16x32, f32 accumulate.
// Weights arrive pre-packed in B-fragment order (w1p, w2p).
// C/D layout: VGPR r -> row r + 8*(lane>>4), col lane&15.
// ---------------------------------------------------------------------------
__global__ __launch_bounds__(256) void block_mlp_kernel(
        float* __restrict__ x, const float* __restrict__ agg, const float* __restrict__ inv,
        const float* __restrict__ ln_g, const float* __restrict__ ln_b,
        const unsigned short* __restrict__ w1p, const float* __restrict__ b1,
        const unsigned short* __restrict__ w2p, const float* __restrict__ b2,
        const float* __restrict__ gamma) {
    __shared__ __align__(16) float          xs[16][H_SZ];
    __shared__ __align__(16) float          as_[16][H_SZ];
    __shared__ __align__(16) float          hf[16][H_SZ];
    __shared__ __align__(16) unsigned short hb[16][H_SZ];
    __shared__ __align__(16) unsigned short act[16][H4_SZ];
    __shared__ float mu_s[16], rs_s[16];

    const int  tid  = threadIdx.x;
    const long base = (long)blockIdx.x * 16;            // first flat token of tile

    __builtin_prefetch(w1p, 0, 1);                      // global_prefetch_b8
    __builtin_prefetch(w2p, 0, 1);

#if defined(HAVE_ASYNC_LDS)
    // stage x and agg tiles (6 KB each, contiguous) via async DMA to LDS
    {
        const float* gx = x   + base * H_SZ;
        const float* ga = agg + base * H_SZ;
        for (int i = tid; i < (16 * H_SZ) / 4; i += 256) {       // 384 x b128 each
            __builtin_amdgcn_global_load_async_to_lds_b128(
                (GLOBAL_AS v4i*)(gx + i * 4), (LDS_AS v4i*)(&xs[0][0] + i * 4), 0, 0);
            __builtin_amdgcn_global_load_async_to_lds_b128(
                (GLOBAL_AS v4i*)(ga + i * 4), (LDS_AS v4i*)(&as_[0][0] + i * 4), 0, 0);
        }
        __builtin_amdgcn_s_wait_asynccnt(0);
    }
    __syncthreads();
    for (int i = tid; i < 16 * H_SZ; i += 256) {
        int t = i / H_SZ, c = i % H_SZ;
        hf[t][c] = (as_[t][c] + xs[t][c]) * inv[(int)((base + t) % N_SZ)];
    }
#else
    for (int i = tid; i < 16 * H_SZ; i += 256) {
        int t = i / H_SZ, c = i % H_SZ;
        long tok = base + t;
        float xv = x[tok * H_SZ + c];
        float av = agg[tok * H_SZ + c];
        xs[t][c] = xv;
        hf[t][c] = (av + xv) * inv[(int)(tok % N_SZ)];
    }
#endif
    __syncthreads();

    // LN statistics (one thread per row)
    if (tid < 16) {
        float s = 0.f, s2 = 0.f;
        for (int c = 0; c < H_SZ; ++c) { float v = hf[tid][c]; s += v; s2 += v * v; }
        float mu  = s * (1.0f / H_SZ);
        float var = s2 * (1.0f / H_SZ) - mu * mu;
        mu_s[tid] = mu;
        rs_s[tid] = rsqrtf(var + LN_EPS);
    }
    __syncthreads();

    // normalize + scale/shift, pack bf16
    for (int i = tid; i < 16 * H_SZ; i += 256) {
        int t = i / H_SZ, c = i % H_SZ;
        float v = (hf[t][c] - mu_s[t]) * rs_s[t] * ln_g[c] + ln_b[c];
        hb[t][c] = f2bf(v);
    }
    __syncthreads();

    const int wave = tid >> 5, lane = tid & 31;
    const int l = lane & 15, hi = lane >> 4;

    // hoist A fragments for K-chunks 0,32,64 (two ds_load_b128 each)
    const v16bf a0 = lds_frag_a(&hb[l][0],  0, hi);
    const v16bf a1 = lds_frag_a(&hb[l][0], 32, hi);
    const v16bf a2 = lds_frag_a(&hb[l][0], 64, hi);

    // GEMM1: 16x96 @ 96x384 -> gelu -> act (each wave does 3 N-tiles)
    #pragma unroll
    for (int tt = 0; tt < 3; ++tt) {
        const int nt = wave + tt * 8;                   // N-tile 0..23
        v8f c = {};
        #pragma unroll
        for (int kc = 0; kc < 3; ++kc) {
            const v16u bu = *(const v16u*)(w1p + (long)(((nt * 3 + kc) * 32) + lane) * 16);
            const v16bf bf = __builtin_bit_cast(v16bf, bu);
            const v16bf af = (kc == 0) ? a0 : (kc == 1) ? a1 : a2;
            c = __builtin_amdgcn_wmma_f32_16x16x32_bf16(false, af, false, bf,
                                                        (short)0, c, false, false);
        }
        #pragma unroll
        for (int r = 0; r < 8; ++r) {
            int row = r + hi * 8, col = nt * 16 + l;
            act[row][col] = f2bf(gelu_exact(c[r] + b1[col]));
        }
    }
    __syncthreads();

    // GEMM2: 16x384 @ 384x96 -> residual (waves 0..5, one N-tile each)
    if (wave < 6) {
        v8f c = {};
        #pragma unroll
        for (int kc = 0; kc < 12; ++kc) {
            const v16bf af = lds_frag_a(&act[l][0], kc * 32, hi);
            const v16u  bu = *(const v16u*)(w2p + (long)(((wave * 12 + kc) * 32) + lane) * 16);
            c = __builtin_amdgcn_wmma_f32_16x16x32_bf16(false, af,
                                                        false, __builtin_bit_cast(v16bf, bu),
                                                        (short)0, c, false, false);
        }
        #pragma unroll
        for (int r = 0; r < 8; ++r) {
            int row = r + hi * 8, col = wave * 16 + l;
            float h2 = c[r] + b2[col];
            x[(base + row) * H_SZ + col] = xs[row][col] + gamma[col] * h2;
        }
    }
}

// ---------------------------------------------------------------------------
// head: LN -> gelu(96->48) -> 48->2        one thread per token
// ---------------------------------------------------------------------------
__global__ __launch_bounds__(128) void head_kernel(
        const float* __restrict__ x, const float* __restrict__ g, const float* __restrict__ bta,
        const float* __restrict__ w1, const float* __restrict__ b1,
        const float* __restrict__ w2, const float* __restrict__ b2,
        float* __restrict__ pred) {
    __shared__ float lnrow[128][H_SZ];
    const int tid = threadIdx.x;
    const long tok = (long)blockIdx.x * 128 + tid;      // B*N divisible by 128

    float s = 0.f, s2 = 0.f;
    const float* xr = x + tok * H_SZ;
    for (int c = 0; c < H_SZ; ++c) { float v = xr[c]; s += v; s2 += v * v; }
    float mu  = s * (1.0f / H_SZ);
    float var = s2 * (1.0f / H_SZ) - mu * mu;
    float rs  = rsqrtf(var + LN_EPS);
    for (int c = 0; c < H_SZ; ++c) lnrow[tid][c] = (xr[c] - mu) * rs * g[c] + bta[c];

    float h1[H_SZ / 2];
    for (int j = 0; j < H_SZ / 2; ++j) {
        float acc = b1[j];
        for (int k = 0; k < H_SZ; ++k) acc += lnrow[tid][k] * w1[k * (H_SZ / 2) + j];
        h1[j] = gelu_exact(acc);
    }
    for (int o = 0; o < OUT_CH; ++o) {
        float acc = b2[o];
        for (int k = 0; k < H_SZ / 2; ++k) acc += h1[k] * w2[k * OUT_CH + o];
        pred[tok * OUT_CH + o] = acc;
    }
}

// ---------------------------------------------------------------------------
// masked compaction: one block per batch row, chunked prefix scan over N
// ---------------------------------------------------------------------------
__global__ __launch_bounds__(256) void compact_kernel(
        const float* __restrict__ pred_all, const int* __restrict__ node_mask,
        float* __restrict__ pred_m, long long* __restrict__ midx,
        unsigned char* __restrict__ vmask) {
    const int bb = blockIdx.x, tid = threadIdx.x;
    const int per = N_SZ / 256;                          // 16 contiguous nodes per thread
    __shared__ int part[256];

    int cnt = 0;
    const int j0 = tid * per;
    for (int k = 0; k < per; ++k) cnt += (node_mask[(long)bb * N_SZ + j0 + k] != 0);
    part[tid] = cnt;
    __syncthreads();
    if (tid == 0) {
        int acc = 0;
        for (int i = 0; i < 256; ++i) { int t = part[i]; part[i] = acc; acc += t; }
    }
    __syncthreads();
    int off = part[tid];
    for (int k = 0; k < per; ++k) {
        int j = j0 + k;
        if (node_mask[(long)bb * N_SZ + j]) {
            long o = (long)bb * N_SZ + off;
            pred_m[o * OUT_CH + 0] = pred_all[((long)bb * N_SZ + j) * OUT_CH + 0];
            pred_m[o * OUT_CH + 1] = pred_all[((long)bb * N_SZ + j) * OUT_CH + 1];
            midx[o]  = (long long)j;
            vmask[o] = 1;
            ++off;
        }
    }
}

// ---------------------------------------------------------------------------
// host driver
// ---------------------------------------------------------------------------
extern "C" void kernel_launch(void* const* d_in, const int* in_sizes, int n_in,
                              void* d_out, int out_size, void* d_ws, size_t ws_size,
                              hipStream_t stream) {
    (void)in_sizes; (void)n_in; (void)out_size; (void)ws_size;

    const float*     nf  = (const float*)d_in[0];
    const float*     ltk = (const float*)d_in[1];
    const int*       nm  = (const int*)d_in[2];
    const long long* ei  = (const long long*)d_in[3];
    const float*     lat_w = (const float*)d_in[4];
    const float*     lat_b = (const float*)d_in[5];
    const float*     in_w  = (const float*)d_in[6];
    const float*     in_b  = (const float*)d_in[7];
    // blocks: d_in[8 + 7*i + {0:ln_g,1:ln_b,2:w1,3:b1,4:w2,5:b2,6:gamma}]
    const float* h_ln_g = (const float*)d_in[29];
    const float* h_ln_b = (const float*)d_in[30];
    const float* h_w1   = (const float*)d_in[31];
    const float* h_b1   = (const float*)d_in[32];
    const float* h_w2   = (const float*)d_in[33];
    const float* h_b2   = (const float*)d_in[34];

    const long ntok = (long)B_SZ * N_SZ;
    const long xcnt = ntok * H_SZ;

    // workspace carve-up (all sizes multiples of 32 B -> fragments stay aligned)
    char* ws = (char*)d_ws;
    float* x    = (float*)ws;                               ws += (size_t)xcnt * 4;
    float* agg  = (float*)ws;                               ws += (size_t)xcnt * 4;
    float* lat  = (float*)ws;                               ws += (size_t)B_SZ * H_SZ * 4;
    float* inv  = (float*)ws;                               ws += (size_t)N_SZ * 4;
    float* deg  = (float*)ws;                               ws += (size_t)N_SZ * 4;
    float* pred = (float*)ws;                               ws += (size_t)ntok * OUT_CH * 4;
    unsigned short* w1p[3]; unsigned short* w2p[3];
    for (int i = 0; i < 3; ++i) { w1p[i] = (unsigned short*)ws; ws += (size_t)H_SZ * H4_SZ * 2; }
    for (int i = 0; i < 3; ++i) { w2p[i] = (unsigned short*)ws; ws += (size_t)H4_SZ * H_SZ * 2; }

    // output carve-up: float pred_masked | int64 mask_indices | bool valid_mask
    float*          pred_m = (float*)d_out;
    long long*      midx   = (long long*)((char*)d_out + (size_t)ntok * OUT_CH * 4);
    unsigned char*  vmask  = (unsigned char*)((char*)midx + (size_t)ntok * 8);
    const long out_bytes = ntok * (OUT_CH * 4 + 8 + 1);

    // weights -> bf16, pre-packed in WMMA B-fragment order
    const int frag1 = (H4_SZ / 16) * (H_SZ / 32) * 32;      // 24 tiles * 3 chunks * 32 lanes
    const int frag2 = (H_SZ / 16) * (H4_SZ / 32) * 32;      // 6 tiles * 12 chunks * 32 lanes
    for (int i = 0; i < 3; ++i) {
        pack_frag_kernel<<<(frag1 + 255) / 256, 256, 0, stream>>>(
            (const float*)d_in[8 + 7 * i + 2], w1p[i], H4_SZ, H4_SZ / 16, H_SZ / 32);
        pack_frag_kernel<<<(frag2 + 255) / 256, 256, 0, stream>>>(
            (const float*)d_in[8 + 7 * i + 4], w2p[i], H_SZ, H_SZ / 16, H4_SZ / 32);
    }

    // latent + input projection
    latent_kernel<<<(B_SZ * H_SZ + 255) / 256, 256, 0, stream>>>(ltk, lat_w, lat_b, lat);
    input_proj_kernel<<<(int)((xcnt + 255) / 256), 256, 0, stream>>>(nf, lat, in_w, in_b, x);

    // degrees
    zero_f32<<<16, 256, 0, stream>>>(deg, N_SZ);
    deg_kernel<<<(E_SZ + 255) / 256, 256, 0, stream>>>(ei, deg);
    inv_kernel<<<(N_SZ + 255) / 256, 256, 0, stream>>>(deg, inv);

    // message-passing blocks
    for (int blk = 0; blk < 3; ++blk) {
        const float* ln_g  = (const float*)d_in[8 + 7 * blk + 0];
        const float* ln_b  = (const float*)d_in[8 + 7 * blk + 1];
        const float* b1    = (const float*)d_in[8 + 7 * blk + 3];
        const float* b2    = (const float*)d_in[8 + 7 * blk + 5];
        const float* gamma = (const float*)d_in[8 + 7 * blk + 6];

        zero_f32<<<4096, 256, 0, stream>>>(agg, xcnt);
        scatter_kernel<<<(int)(((long)B_SZ * E_SZ + 255) / 256), 256, 0, stream>>>(x, ei, agg);
        block_mlp_kernel<<<(int)(ntok / 16), 256, 0, stream>>>(
            x, agg, inv, ln_g, ln_b, w1p[blk], b1, w2p[blk], b2, gamma);
    }

    // head + masked compaction
    head_kernel<<<(int)(ntok / 128), 128, 0, stream>>>(x, h_ln_g, h_ln_b, h_w1, h_b1, h_w2, h_b2, pred);
    zero_bytes<<<8192, 256, 0, stream>>>((unsigned char*)d_out, out_bytes);
    compact_kernel<<<B_SZ, 256, 0, stream>>>(pred, nm, pred_m, midx, vmask);
}
